// HGCLayer_77403900608996
// MI455X (gfx1250) — matrix operand order repaired and checked by reading, not compiled, not run on hardware
//
#include <hip/hip_runtime.h>

typedef __attribute__((ext_vector_type(16))) _Float16 v16h;
typedef __attribute__((ext_vector_type(8)))  float    v8f;

#define NHEAD 8
#define HDIM 16
#define OCH 128

#ifndef __has_builtin
#define __has_builtin(x) 0
#endif
#if __has_builtin(__builtin_amdgcn_tensor_load_to_lds) && __has_builtin(__builtin_amdgcn_s_wait_tensorcnt)
#define USE_TDM 1
#else
#define USE_TDM 0
#endif

// ---------- helpers ----------
static __device__ __forceinline__ unsigned encf(float f) {
    unsigned u = __float_as_uint(f);
    return (u & 0x80000000u) ? ~u : (u | 0x80000000u);   // monotone float->uint
}
static __device__ __forceinline__ float decf(unsigned u) {
    unsigned v = (u & 0x80000000u) ? (u & 0x7FFFFFFFu) : ~u;
    return __uint_as_float(v);
}

#if USE_TDM
typedef __attribute__((ext_vector_type(4))) unsigned tdm_v4u;
typedef __attribute__((ext_vector_type(8))) int      tdm_v8i;
typedef __attribute__((ext_vector_type(4))) int      tdm_v4i;

// Issue one TDM 2D tile load: rows_remaining x 128 f16 elements (row stride 128)
// from gbase into LDS at lds_addr; rows beyond rows_remaining are zero-filled
// by the TDM OOB rule; tile is 128x128.
static __device__ __forceinline__ void tdm_load_tile_2d(const _Float16* gbase,
                                                        unsigned lds_addr,
                                                        unsigned rows_remaining) {
    unsigned long long ga = (unsigned long long)(uintptr_t)gbase;
    tdm_v4u g0;
    g0[0] = 1u;                                            // count=1, no gather
    g0[1] = lds_addr;                                      // lds_addr bits[63:32]
    g0[2] = (unsigned)(ga & 0xFFFFFFFFull);                // global_addr lo
    g0[3] = (unsigned)((ga >> 32) & 0x01FFFFFFull) | 0x80000000u;  // ga hi | type=2
    tdm_v8i g1;
    g1[0] = 0x00010000;                                    // mask=0, data_size=1 (2B)
    g1[1] = (int)(128u << 16);                             // tensor_dim0 = 128 (lo16 in [31:16])
    g1[2] = (int)((rows_remaining & 0xFFFFu) << 16);       // tensor_dim1 lo16
    g1[3] = (int)(((rows_remaining >> 16) & 0xFFFFu) | (128u << 16)); // dim1 hi | tile_dim0=128
    g1[4] = 128;                                           // tile_dim1 = 128 rows, tile_dim2 = 0
    g1[5] = 128;                                           // tensor_dim0_stride = 128
    g1[6] = 0;
    g1[7] = 0;
    tdm_v4i gz = {0, 0, 0, 0};
#if defined(__clang_major__) && (__clang_major__ >= 23)
    tdm_v8i gz8 = {0, 0, 0, 0, 0, 0, 0, 0};
    __builtin_amdgcn_tensor_load_to_lds(g0, g1, gz, gz, gz8, 0);
#else
    __builtin_amdgcn_tensor_load_to_lds(g0, g1, gz, gz, 0);
#endif
}
#endif

// ---------- small prep kernels ----------
__global__ void cvt_f16_kernel(const float* __restrict__ src, _Float16* __restrict__ dst, long n) {
    long i = (long)blockIdx.x * blockDim.x + threadIdx.x;
    if (i < n) dst[i] = (_Float16)src[i];
}

// w: [2][128 out][128 in] f32  ->  wt: [2][128 in][128 out] f16
__global__ void wt_transpose_kernel(const float* __restrict__ w, _Float16* __restrict__ wt) {
    int i = blockIdx.x * blockDim.x + threadIdx.x;       // 2*128*128
    int t = i >> 14, o = (i >> 7) & 127, c = i & 127;
    wt[t * 16384 + c * 128 + o] = (_Float16)w[i];
}

// kw2t[i_in][o] = sum_d a_rel[e][h(o)][d][eo(o)] * k_w[st][h*16+d][i_in]   (already transposed)
__global__ void fuse_rel_kernel(const float* __restrict__ kw, const float* __restrict__ arel,
                                _Float16* __restrict__ kw2t, int e, int st) {
    int idx = blockIdx.x * blockDim.x + threadIdx.x;     // 128*128 : o*128 + i
    int o = idx >> 7, ii = idx & 127;
    int h = o >> 4, eo = o & 15;
    float s = 0.f;
#pragma unroll
    for (int d = 0; d < 16; ++d)
        s += arel[((e * NHEAD + h) * 16 + d) * 16 + eo] * kw[(st * 128 + h * 16 + d) * 128 + ii];
    kw2t[ii * 128 + o] = (_Float16)s;
}

__global__ void fuse_bias_kernel(const float* __restrict__ kb, const float* __restrict__ arel,
                                 float* __restrict__ bout, int e, int st) {
    int o = blockIdx.x * blockDim.x + threadIdx.x;
    if (o >= 128) return;
    int h = o >> 4, eo = o & 15;
    float s = 0.f;
#pragma unroll
    for (int d = 0; d < 16; ++d)
        s += arel[((e * NHEAD + h) * 16 + d) * 16 + eo] * kb[st * 128 + h * 16 + d];
    bout[o] = s;
}

__global__ void zero_f32_kernel(float* p, long n) {
    long i = (long)blockIdx.x * blockDim.x + threadIdx.x;
    if (i < n) p[i] = 0.f;
}
__global__ void zero_u32_kernel(unsigned* p, long n) {
    long i = (long)blockIdx.x * blockDim.x + threadIdx.x;
    if (i < n) p[i] = 0u;   // encodes "below every finite encf(x)"
}

// ---------- shared GEMM plumbing ----------
// Stage a 128x128 A tile (row rbase..rbase+127, zero-padded past nrows) and the
// full 128x128 transposed weight into LDS.
static __device__ __forceinline__ void stage_tiles(const _Float16* __restrict__ A,
                                                   const _Float16* __restrict__ Wt,
                                                   _Float16* As, _Float16* Bs,
                                                   int rbase, int nrows, int tid) {
#if USE_TDM
    if (tid == 0) {
        unsigned rem = (rbase < nrows) ? (unsigned)(nrows - rbase) : 0u;
        tdm_load_tile_2d(A + (long)rbase * 128, (unsigned)(uintptr_t)As, rem);
        tdm_load_tile_2d(Wt, (unsigned)(uintptr_t)Bs, 128u);
        __builtin_amdgcn_s_wait_tensorcnt(0);
    }
    __syncthreads();
#else
    for (int idx = tid; idx < 128 * 16; idx += 256) {     // 16B chunks
        int r = idx >> 4, c8 = idx & 15;
        *(reinterpret_cast<uint4*>(Bs + r * 128) + c8) =
            *(reinterpret_cast<const uint4*>(Wt + r * 128) + c8);
        int gr = rbase + r;
        uint4 va = make_uint4(0, 0, 0, 0);
        if (gr < nrows) va = *(reinterpret_cast<const uint4*>(A + (long)gr * 128) + c8);
        *(reinterpret_cast<uint4*>(As + r * 128) + c8) = va;
    }
    __syncthreads();
#endif
}

// Core 16-row-strip x 128-col WMMA accumulation for one wave.
static __device__ __forceinline__ void wmma_strip(const _Float16* As, const _Float16* Bs,
                                                  int r0, int lane, v8f acc[8]) {
    int m = lane & 15;
    int kb = (lane < 16) ? 0 : 8;                        // A-frag K split per ISA layout
    for (int kc = 0; kc < 128; kc += 32) {
        v16h af;
        const _Float16* arow = As + (r0 + m) * 128 + kc + kb;
        reinterpret_cast<uint4*>(&af)[0] = *reinterpret_cast<const uint4*>(arow);       // K=kb..kb+7
        reinterpret_cast<uint4*>(&af)[1] = *reinterpret_cast<const uint4*>(arow + 16);  // K=kb+16..kb+23
        const _Float16* brow = Bs + kc * 128 + m;         // B-frag: lane = N col, K split same as A
        v16h bf[8];
#pragma unroll
        for (int nt = 0; nt < 8; ++nt) {                 // preload all 8 B frags -> batched ds loads
#pragma unroll
            for (int s = 0; s < 8; ++s) {
                bf[nt][s]     = brow[(kb + s) * 128 + nt * 16];
                bf[nt][8 + s] = brow[(16 + kb + s) * 128 + nt * 16];
            }
        }
#pragma unroll
        for (int nt = 0; nt < 8; ++nt)
            acc[nt] = __builtin_amdgcn_wmma_f32_16x16x32_f16(false, af, false, bf[nt],
                                                             (short)0, acc[nt], false, false);
    }
}

// ---------- WMMA GEMM: C[nrows x 128] (f16) = A[nrows x 128] (f16) @ Wt + bias ----------
__global__ __launch_bounds__(256) void gemm_f16_kernel(
    const _Float16* __restrict__ A, const _Float16* __restrict__ Wt,
    const float* __restrict__ bias, _Float16* __restrict__ C, int nrows) {
    __shared__ _Float16 As[128 * 128];
    __shared__ _Float16 Bs[128 * 128];
    int tid = threadIdx.x;
    int rbase = blockIdx.x * 128;
    stage_tiles(A, Wt, As, Bs, rbase, nrows, tid);

    int wave = tid >> 5, lane = tid & 31;
    int r0 = wave * 16;
    v8f zv = {0.f, 0.f, 0.f, 0.f, 0.f, 0.f, 0.f, 0.f};
    v8f acc[8];
#pragma unroll
    for (int nt = 0; nt < 8; ++nt) acc[nt] = zv;
    wmma_strip(As, Bs, r0, lane, acc);

    int col = lane & 15;
    int rsel = (lane >> 4) * 8;                          // D layout: lane=N, VGPR=M
#pragma unroll
    for (int nt = 0; nt < 8; ++nt) {
        float b = bias ? bias[nt * 16 + col] : 0.f;
#pragma unroll
        for (int j = 0; j < 8; ++j) {
            int grow = rbase + r0 + rsel + j;
            if (grow < nrows)
                C[(long)grow * 128 + nt * 16 + col] = (_Float16)(acc[nt][j] + b);
        }
    }
}

// ---------- final WMMA GEMM with fused bias + sigmoid-skip + ReLU + LayerNorm ----------
__global__ __launch_bounds__(256) void gemm_final_kernel(
    const _Float16* __restrict__ A, const _Float16* __restrict__ Wt,
    const float* __restrict__ bias, const float* __restrict__ xres,
    const float* __restrict__ skipPtr, const float* __restrict__ ln_g,
    const float* __restrict__ ln_b, float* __restrict__ out, int nrows) {
    __shared__ _Float16 As[128 * 128];
    __shared__ _Float16 Bs[128 * 128];
    int tid = threadIdx.x;
    int rbase = blockIdx.x * 128;
    stage_tiles(A, Wt, As, Bs, rbase, nrows, tid);

    int wave = tid >> 5, lane = tid & 31;
    int r0 = wave * 16;
    v8f zv = {0.f, 0.f, 0.f, 0.f, 0.f, 0.f, 0.f, 0.f};
    v8f acc[8];
#pragma unroll
    for (int nt = 0; nt < 8; ++nt) acc[nt] = zv;
    wmma_strip(As, Bs, r0, lane, acc);

    float beta = 1.f / (1.f + __expf(-skipPtr[0]));
    int col = lane & 15;
    int rsel = (lane >> 4) * 8;
#pragma unroll
    for (int j = 0; j < 8; ++j) {
        int grow = rbase + r0 + rsel + j;
        bool ok = grow < nrows;
        float ov[8];
        float s = 0.f, sq = 0.f;
#pragma unroll
        for (int nt = 0; nt < 8; ++nt) {
            int c = nt * 16 + col;
            float o = acc[nt][j] + bias[c];
            float xv = ok ? xres[(long)grow * 128 + c] : 0.f;
            o = beta * o + (1.f - beta) * xv;
            o = fmaxf(o, 0.f);
            ov[nt] = o;
            s += o; sq += o * o;
        }
        // reduce across the 16 lanes holding this row (xor masks 1..8 stay inside a half-wave)
#pragma unroll
        for (int off = 1; off < 16; off <<= 1) {
            s  += __shfl_xor(s,  off);
            sq += __shfl_xor(sq, off);
        }
        float mu  = s * (1.f / 128.f);
        float var = sq * (1.f / 128.f) - mu * mu;
        float rs  = rsqrtf(var + 1e-5f);
        if (ok) {
#pragma unroll
            for (int nt = 0; nt < 8; ++nt) {
                int c = nt * 16 + col;
                out[(long)grow * 128 + c] = (ov[nt] - mu) * rs * ln_g[c] + ln_b[c];
            }
        }
    }
}

// ---------- edge phase ----------
// one thread per (edge, head): logit = (q[dst,h,:] . ke[src,h,:]) * p_rel[h] / sqrt(16)
__global__ void alpha_kernel(const _Float16* __restrict__ Q, const _Float16* __restrict__ KE,
                             const int* __restrict__ src, const int* __restrict__ dst,
                             const float* __restrict__ prel, float* __restrict__ logits,
                             unsigned* __restrict__ segmax, long EH) {
    long i = (long)blockIdx.x * blockDim.x + threadIdx.x;
    if (i >= EH) return;
    long eidx = i >> 3;
    int h = (int)(i & 7);
    int sv = src[eidx], dv = dst[eidx];
    const _Float16* qp = Q + (long)dv * 128 + h * 16;
    const _Float16* kp = KE + (long)sv * 128 + h * 16;
    v16h qv, kv;
    reinterpret_cast<uint4*>(&qv)[0] = reinterpret_cast<const uint4*>(qp)[0];
    reinterpret_cast<uint4*>(&qv)[1] = reinterpret_cast<const uint4*>(qp)[1];
    reinterpret_cast<uint4*>(&kv)[0] = reinterpret_cast<const uint4*>(kp)[0];
    reinterpret_cast<uint4*>(&kv)[1] = reinterpret_cast<const uint4*>(kp)[1];
    float acc = 0.f;
#pragma unroll
    for (int d = 0; d < 16; ++d) acc += (float)qv[d] * (float)kv[d];
    float logit = acc * prel[h] * 0.25f;   // 1/sqrt(16)
    logits[i] = logit;
    atomicMax(segmax + (long)dv * NHEAD + h, encf(logit));
}

__global__ void exp_kernel(float* __restrict__ logits, const int* __restrict__ dst,
                           const unsigned* __restrict__ segmax, float* __restrict__ segsum, long EH) {
    long i = (long)blockIdx.x * blockDim.x + threadIdx.x;
    if (i >= EH) return;
    long eidx = i >> 3;
    int h = (int)(i & 7);
    int dv = dst[eidx];
    float mx = decf(segmax[(long)dv * NHEAD + h]);
    if (!(mx > -3.0e38f && mx < 3.0e38f)) mx = 0.f;      // mirror reference isfinite guard
    float ex = __expf(logits[i] - mx);
    logits[i] = ex;
    unsafeAtomicAdd(segsum + (long)dv * NHEAD + h, ex);
}

// one thread per (edge, channel): agg[dst,ch] += ve[src,ch] * ex/(den+eps)
__global__ void msg_kernel(const _Float16* __restrict__ VE, const float* __restrict__ ex,
                           const float* __restrict__ segsum, const int* __restrict__ src,
                           const int* __restrict__ dst, float* __restrict__ agg, long EC) {
    long i = (long)blockIdx.x * blockDim.x + threadIdx.x;
    if (i >= EC) return;
    long eidx = i >> 7;
    int ch = (int)(i & 127);
    int h = ch >> 4;
    int sv = src[eidx], dv = dst[eidx];
    float w = ex[eidx * NHEAD + h] / (segsum[(long)dv * NHEAD + h] + 1e-16f);
    float val = (float)VE[(long)sv * 128 + ch] * w;
    unsafeAtomicAdd(agg + (long)dv * 128 + ch, val);
}

__global__ void gelu_kernel(const float* __restrict__ agg, _Float16* __restrict__ gh, long n) {
    long i = (long)blockIdx.x * blockDim.x + threadIdx.x;
    if (i >= n) return;
    float a = agg[i];
    float g = 0.5f * a * (1.f + erff(a * 0.70710678118654752f));
    gh[i] = (_Float16)g;
}

// ---------- launcher ----------
static inline int nblk(long n, int b) { return (int)((n + b - 1) / b); }

extern "C" void kernel_launch(void* const* d_in, const int* in_sizes, int n_in,
                              void* d_out, int out_size, void* d_ws, size_t ws_size,
                              hipStream_t stream) {
    (void)n_in; (void)out_size; (void)ws_size;
    const float* x     = (const float*)d_in[0];
    const float* k_w   = (const float*)d_in[1];
    const float* k_b   = (const float*)d_in[2];
    const float* q_w   = (const float*)d_in[3];
    const float* q_b   = (const float*)d_in[4];
    const float* v_w   = (const float*)d_in[5];
    const float* v_b   = (const float*)d_in[6];
    const float* a_w   = (const float*)d_in[7];
    const float* a_b   = (const float*)d_in[8];
    const float* skip  = (const float*)d_in[9];
    const float* a_rel = (const float*)d_in[10];
    const float* m_rel = (const float*)d_in[11];
    const float* p_rel = (const float*)d_in[12];
    const float* ln_g  = (const float*)d_in[13];
    const float* ln_b  = (const float*)d_in[14];
    const int*   edge  = (const int*)d_in[15];

    long N = (long)in_sizes[0] / 256;     // x: [2, N, 128]
    long E = (long)in_sizes[15] / 4;      // edge_index: [2, 2, E]
    size_t NTe = (size_t)N * 128;         // elements per (type) feature table

    // workspace carve (256B aligned)
    char* ws = (char*)d_ws;
    auto carve = [&](size_t bytes) -> char* {
        char* p = ws;
        ws += (bytes + 255) & ~(size_t)255;
        return p;
    };
    _Float16* xh    = (_Float16*)carve(2 * NTe * sizeof(_Float16));
    _Float16* Qh    = (_Float16*)carve(2 * NTe * sizeof(_Float16));
    _Float16* KEh   = (_Float16*)carve(2 * NTe * sizeof(_Float16));
    _Float16* VEh   = (_Float16*)carve(2 * NTe * sizeof(_Float16));
    _Float16* Gh    = (_Float16*)carve(2 * NTe * sizeof(_Float16));
    float*    agg   = (float*)   carve(2 * NTe * sizeof(float));
    _Float16* qwt   = (_Float16*)carve(2 * 16384 * sizeof(_Float16));
    _Float16* awt   = (_Float16*)carve(2 * 16384 * sizeof(_Float16));
    _Float16* kw2t  = (_Float16*)carve(2 * 16384 * sizeof(_Float16));
    _Float16* vw2t  = (_Float16*)carve(2 * 16384 * sizeof(_Float16));
    float*    keb   = (float*)   carve(2 * 128 * sizeof(float));
    float*    vbe   = (float*)   carve(2 * 128 * sizeof(float));
    float*    logits= (float*)   carve((size_t)E * NHEAD * sizeof(float));
    unsigned* segmax= (unsigned*)carve((size_t)N * NHEAD * sizeof(unsigned));
    float*    segsum= (float*)   carve((size_t)N * NHEAD * sizeof(float));

    const int B = 256;
    // 1) f16 conversions + weight prep
    cvt_f16_kernel<<<nblk(2 * (long)NTe, B), B, 0, stream>>>(x, xh, 2 * (long)NTe);
    wt_transpose_kernel<<<nblk(2 * 16384, B), B, 0, stream>>>(q_w, qwt);
    wt_transpose_kernel<<<nblk(2 * 16384, B), B, 0, stream>>>(a_w, awt);
    // EDGE_META: e=0 -> (src=0,dst=1), e=1 -> (src=1,dst=0)
    fuse_rel_kernel<<<nblk(16384, B), B, 0, stream>>>(k_w, a_rel, kw2t + 0 * 16384, 0, 0);
    fuse_rel_kernel<<<nblk(16384, B), B, 0, stream>>>(k_w, a_rel, kw2t + 1 * 16384, 1, 1);
    fuse_rel_kernel<<<nblk(16384, B), B, 0, stream>>>(v_w, m_rel, vw2t + 0 * 16384, 0, 0);
    fuse_rel_kernel<<<nblk(16384, B), B, 0, stream>>>(v_w, m_rel, vw2t + 1 * 16384, 1, 1);
    fuse_bias_kernel<<<1, 128, 0, stream>>>(k_b, a_rel, keb + 0 * 128, 0, 0);
    fuse_bias_kernel<<<1, 128, 0, stream>>>(k_b, a_rel, keb + 1 * 128, 1, 1);
    fuse_bias_kernel<<<1, 128, 0, stream>>>(v_b, m_rel, vbe + 0 * 128, 0, 0);
    fuse_bias_kernel<<<1, 128, 0, stream>>>(v_b, m_rel, vbe + 1 * 128, 1, 1);

    // 2) WMMA GEMMs: Q[t], KE[e], VE[e]
    int gGemm = nblk(N, 128);
    for (int t = 0; t < 2; ++t)
        gemm_f16_kernel<<<gGemm, B, 0, stream>>>(xh + t * NTe, qwt + t * 16384,
                                                 q_b + t * 128, Qh + t * NTe, (int)N);
    const int stOf[2] = {0, 1};
    for (int e = 0; e < 2; ++e) {
        gemm_f16_kernel<<<gGemm, B, 0, stream>>>(xh + stOf[e] * NTe, kw2t + e * 16384,
                                                 keb + e * 128, KEh + e * NTe, (int)N);
        gemm_f16_kernel<<<gGemm, B, 0, stream>>>(xh + stOf[e] * NTe, vw2t + e * 16384,
                                                 vbe + e * 128, VEh + e * NTe, (int)N);
    }

    // 3) edge phase
    zero_f32_kernel<<<nblk(2 * (long)NTe, B), B, 0, stream>>>(agg, 2 * (long)NTe);
    const int dtOf[2] = {1, 0};
    for (int e = 0; e < 2; ++e) {
        const int* srcp = edge + ((long)e * 2 + 0) * E;
        const int* dstp = edge + ((long)e * 2 + 1) * E;
        long NH = N * NHEAD, EH = E * NHEAD, EC = E * 128;
        zero_u32_kernel<<<nblk(NH, B), B, 0, stream>>>(segmax, NH);
        zero_f32_kernel<<<nblk(NH, B), B, 0, stream>>>(segsum, NH);
        alpha_kernel<<<nblk(EH, B), B, 0, stream>>>(Qh + dtOf[e] * NTe, KEh + e * NTe,
                                                    srcp, dstp, p_rel + e * NHEAD,
                                                    logits, segmax, EH);
        exp_kernel<<<nblk(EH, B), B, 0, stream>>>(logits, dstp, segmax, segsum, EH);
        msg_kernel<<<nblk(EC, B), B, 0, stream>>>(VEh + e * NTe, logits, segsum,
                                                  srcp, dstp, agg + dtOf[e] * NTe, EC);
    }

    // 4) GELU + final fused GEMM (bias + skip + ReLU + LayerNorm)
    gelu_kernel<<<nblk(2 * (long)NTe, B), B, 0, stream>>>(agg, Gh, 2 * (long)NTe);
    for (int t = 0; t < 2; ++t)
        gemm_final_kernel<<<gGemm, B, 0, stream>>>(Gh + t * NTe, awt + t * 16384,
                                                   a_b + t * 128, x + t * NTe,
                                                   skip + t, ln_g, ln_b,
                                                   (float*)d_out + t * NTe, (int)N);
}
